// LinearQ4_0_5282809775038
// MI455X (gfx1250) — compile-verified, matrix operand order checked by reading, not run on hardware
//
#include <hip/hip_runtime.h>
#include <stdint.h>

typedef _Float16 v16h __attribute__((ext_vector_type(16)));
typedef _Float16 v8h  __attribute__((ext_vector_type(8)));
typedef _Float16 v2h  __attribute__((ext_vector_type(2)));
typedef float    v8f  __attribute__((ext_vector_type(8)));
typedef uint32_t u32x4 __attribute__((ext_vector_type(4)));
typedef uint32_t u32x8 __attribute__((ext_vector_type(8)));

#define IN_F  4096
#define OUT_F 11008
#define M_TOT 8192

#define BM 64
#define BN 128
#define BK 128
#define LDS_K 136          // 128 + 8-halves pad -> 272B stride, conflict-free
#define NT   (IN_F / BK)   // 32 K-steps

// One-shot TDM: move a 64-row x 256B 2D tile (row stride 8192B) global->LDS,
// inserting 16B of LDS padding after every 256B row (pad_interval=5: 64 DWORDs,
// pad_amount=3: 4 DWORDs). data_size=8B units. Issued by one wave, TENSORcnt.
__device__ __forceinline__ void tdm_load_x_tile(uint32_t lds_addr, uint64_t ga)
{
    u32x4 g0;
    g0[0] = 1u;                                   // count=1, user D#, no gather
    g0[1] = lds_addr;                             // LDS byte address of tile
    g0[2] = (uint32_t)ga;                         // global_addr[31:0]
    g0[3] = (uint32_t)((ga >> 32) & 0x01FFFFFFu)  // global_addr[56:32]
          | 0x80000000u;                          // type=2 ("image")

    u32x8 g1;
    g1[0] = (3u << 16)                            // data_size = 8B
          | (1u << 20)                            // pad_enable
          | (5u << 22)                            // pad_interval: 64 DWORDs (256B)
          | (3u << 25);                           // pad_amount: 4 DWORDs (16B)
    g1[1] = 32u << 16;                            // tensor_dim0 = 32 units (256B)
    g1[2] = 64u << 16;                            // tensor_dim0 hi | tensor_dim1 = 64
    g1[3] = 32u << 16;                            // tensor_dim1 hi | tile_dim0 = 32
    g1[4] = 64u;                                  // tile_dim1 = 64, tile_dim2 = 0
    g1[5] = 1024u;                                // tensor_dim0_stride = 8192B / 8
    g1[6] = 0u;
    g1[7] = 0u;

    asm volatile("tensor_load_to_lds %0, %1" :: "s"(g0), "s"(g1) : "memory");
}

// Two fp16 dequants in one shot: bits = two lanes of (0x6400 | (u^8)) which is
// fp16(1032 + s) exactly; result = bits*s_scale - 1032*s_scale  (v_pk_fma_f16)
__device__ __forceinline__ uint32_t fma2_u32(uint32_t bits, v2h ss, v2h bb)
{
    v2h v = __builtin_bit_cast(v2h, bits);
    v = v * ss + bb;
    return __builtin_bit_cast(uint32_t, v);
}

// Dequantize this thread's 32 packed bytes (q[0..7]) into 64 fp16 values and
// commit to the LDS W-tile. hi nibble -> K j, lo nibble -> K 64+j.
__device__ __forceinline__ void dequant_commit(_Float16* bd, int cb,
                                               const uint32_t* q, v2h s2)
{
    const _Float16 nkh = (_Float16)(-1032.0f);
    const v2h ss0 = { s2[0], s2[0] };
    const v2h ss1 = { s2[1], s2[1] };
    const v2h bb0 = { nkh * s2[0], nkh * s2[0] };
    const v2h bb1 = { nkh * s2[1], nkh * s2[1] };

    uint32_t hbuf[16], lbuf[16];
    #pragma unroll
    for (int c = 0; c < 8; ++c) {
        uint32_t w  = q[c];
        uint32_t hn = ((w >> 4) & 0x0F0F0F0Fu) ^ 0x08080808u;   // (u^8) hi nibbles
        uint32_t ln = ( w       & 0x0F0F0F0Fu) ^ 0x08080808u;   // (u^8) lo nibbles
        // v_perm_b32: splice 0x64 exponent bytes around nibble bytes -> 2 fp16/op
        hbuf[2*c]   = fma2_u32(__builtin_amdgcn_perm(0x64646464u, hn, 0x05010400u), ss0, bb0);
        hbuf[2*c+1] = fma2_u32(__builtin_amdgcn_perm(0x64646464u, hn, 0x05030402u), ss0, bb0);
        lbuf[2*c]   = fma2_u32(__builtin_amdgcn_perm(0x64646464u, ln, 0x05010400u), ss1, bb1);
        lbuf[2*c+1] = fma2_u32(__builtin_amdgcn_perm(0x64646464u, ln, 0x05030402u), ss1, bb1);
    }
    #pragma unroll
    for (int c = 0; c < 4; ++c) {
        *(uint4*)(bd + cb + 8 * c)      = *(uint4*)&hbuf[4 * c];
        *(uint4*)(bd + 64 + cb + 8 * c) = *(uint4*)&lbuf[4 * c];
    }
}

__global__ __launch_bounds__(256)
void q40_gemm_wmma(const _Float16* __restrict__ x,
                   const uint8_t*  __restrict__ wq,
                   const _Float16* __restrict__ sc,
                   float* __restrict__ out)
{
    __shared__ alignas(16) _Float16 As[2][BM][LDS_K];   // 2 x 17.0 KB, TDM-filled
    __shared__ alignas(16) _Float16 Bs[2][BN][LDS_K];   // 2 x 34.0 KB, dequantized W

    const int t    = threadIdx.x;
    const int lane = t & 31;
    const int wave = t >> 5;        // 0..7
    const int wm   = wave >> 2;     // 0..1 : 32-row M strip
    const int wn   = wave & 3;      // 0..3 : 32-col N strip

    const int m0 = blockIdx.x * BM; // 128 M-blocks
    const int n0 = blockIdx.y * BN; //  86 N-blocks

    // W staging: 128 rows x 64 packed bytes/tile, each thread owns 32 bytes
    const int b_row = t >> 1;
    const int b_cb  = (t & 1) * 32;

    const int row = lane & 15;      // M (A) / N (B) index within 16
    const int hi  = lane >> 4;      // K-octet selector per ISA frag layout

    v8f acc00 = {}, acc01 = {}, acc10 = {}, acc11 = {};

    // LDS byte addresses (low 32 bits of flat address, ISA 10.2) - uniform
    const uint32_t a_lds0 = (uint32_t)(uintptr_t)&As[0][0][0];
    const uint32_t a_lds1 = (uint32_t)(uintptr_t)&As[1][0][0];
    const uint64_t x_tile0 = (uint64_t)(uintptr_t)(x + (size_t)m0 * IN_F);

    const int o = n0 + b_row;                       // weight row (output feature)
    const uint8_t*  w_gsrc = wq + (size_t)o * 2048; // o * 32 rows * 64 B
    const _Float16* s_gsrc = sc + (size_t)o * 64;

    uint32_t q[8];
    v2h s2;

    // ---------------- prologue: fetch + commit tile 0 -> buffer 0 ----
    if (wave == 0)
        tdm_load_x_tile(a_lds0, x_tile0);
    {
        const uint8_t* srcw = w_gsrc + b_cb;
        *(uint4*)(&q[0]) = *(const uint4*)(srcw);
        *(uint4*)(&q[4]) = *(const uint4*)(srcw + 16);
        s2 = *(const v2h*)(s_gsrc);
        dequant_commit(&Bs[0][b_row][0], b_cb, q, s2);
    }

    int p = 0;
    for (int kt = 0; kt < NT; ++kt) {
        if (wave == 0)
            __builtin_amdgcn_s_wait_tensorcnt(0);   // TDM tile p landed
        __syncthreads();                            // publish buffer p

        const bool more = (kt + 1 < NT);

        // ---- issue next tile's fetches (latency overlapped) ---------
        if (more) {
            if (wave == 0)
                tdm_load_x_tile(p ? a_lds0 : a_lds1,
                                x_tile0 + (size_t)(kt + 1) * (BK * 2));

            const uint8_t* srcw = w_gsrc + (size_t)(kt + 1) * 64 + b_cb;
            *(uint4*)(&q[0]) = *(const uint4*)(srcw);
            *(uint4*)(&q[4]) = *(const uint4*)(srcw + 16);
            s2 = *(const v2h*)(s_gsrc + 2 * (kt + 1));
            if (kt + 2 < NT)
                __builtin_prefetch(srcw + 64, 0, 1);   // global_prefetch_b8
        }

        // ---- compute on buffer p: 2x2 register tile, 16 WMMAs -------
        {
            const _Float16* a_base0 = &As[p][32 * wm + row][0];
            const _Float16* a_base1 = &As[p][32 * wm + 16 + row][0];
            const _Float16* b_base0 = &Bs[p][32 * wn + row][0];
            const _Float16* b_base1 = &Bs[p][32 * wn + 16 + row][0];

            #pragma unroll
            for (int kk = 0; kk < BK; kk += 32) {
                const int klo = kk + 8 * hi;        // K 0-7 / 8-15 octet
                const int khi = kk + 16 + 8 * hi;   // K 16-23 / 24-31 octet

                v16h a0, a1, b0, b1;
                ((v8h*)&a0)[0] = *(const v8h*)(a_base0 + klo);
                ((v8h*)&a0)[1] = *(const v8h*)(a_base0 + khi);
                ((v8h*)&a1)[0] = *(const v8h*)(a_base1 + klo);
                ((v8h*)&a1)[1] = *(const v8h*)(a_base1 + khi);
                ((v8h*)&b0)[0] = *(const v8h*)(b_base0 + klo);
                ((v8h*)&b0)[1] = *(const v8h*)(b_base0 + khi);
                ((v8h*)&b1)[0] = *(const v8h*)(b_base1 + klo);
                ((v8h*)&b1)[1] = *(const v8h*)(b_base1 + khi);

                acc00 = __builtin_amdgcn_wmma_f32_16x16x32_f16(
                            false, a0, false, b0, (short)0, acc00, false, false);
                acc01 = __builtin_amdgcn_wmma_f32_16x16x32_f16(
                            false, a0, false, b1, (short)0, acc01, false, false);
                acc10 = __builtin_amdgcn_wmma_f32_16x16x32_f16(
                            false, a1, false, b0, (short)0, acc10, false, false);
                acc11 = __builtin_amdgcn_wmma_f32_16x16x32_f16(
                            false, a1, false, b1, (short)0, acc11, false, false);
            }
        }

        // ---- dequant + commit next W tile into buffer p^1 -----------
        // (after compute's ds_loads: DS pipe is in-order within a wave)
        if (more)
            dequant_commit(&Bs[p ^ 1][b_row][0], b_cb, q, s2);

        p ^= 1;
    }

    // ---- store D per C/D VGPR layout (VGPR r -> M = 8*hi + r) -------
    const int colb  = n0 + 32 * wn + row;
    const int rowb0 = m0 + 32 * wm + 8 * hi;
    const int rowb1 = rowb0 + 16;
    #pragma unroll
    for (int r = 0; r < 8; ++r) {
        float* o0 = out + (size_t)(rowb0 + r) * OUT_F + colb;
        float* o1 = out + (size_t)(rowb1 + r) * OUT_F + colb;
        o0[0]  = acc00[r];
        o0[16] = acc01[r];
        o1[0]  = acc10[r];
        o1[16] = acc11[r];
    }
}

extern "C" void kernel_launch(void* const* d_in, const int* in_sizes, int n_in,
                              void* d_out, int out_size, void* d_ws, size_t ws_size,
                              hipStream_t stream) {
    (void)in_sizes; (void)n_in; (void)out_size; (void)d_ws; (void)ws_size;
    const _Float16* x  = (const _Float16*)d_in[0];
    const uint8_t*  wq = (const uint8_t*)d_in[1];
    const _Float16* sc = (const _Float16*)d_in[2];
    float* out = (float*)d_out;

    dim3 grid(M_TOT / BM, OUT_F / BN);   // 128 x 86
    q40_gemm_wmma<<<grid, dim3(256), 0, stream>>>(x, wq, sc, out);
}
